// HolographicCodeGNN_45449343926754
// MI455X (gfx1250) — matrix operand order repaired and checked by reading, not compiled, not run on hardware
//
#include <hip/hip_runtime.h>
#include <math.h>

#define NN   200
#define HH   128
#define DD   8
#define LL   6
#define EE   16384
#define SSZ  8
#define FFD  512
#define MP   208     // 200 padded to 13*16
#define MTILES 13

typedef __attribute__((ext_vector_type(16))) __bf16  v16bf;
typedef __attribute__((ext_vector_type(8)))  float   v8f;

__device__ __forceinline__ float geluf(float x) {
  return 0.5f * x * (1.0f + erff(x * 0.70710678118654752f));
}
__device__ __forceinline__ float wred_sum(float v) {
  #pragma unroll
  for (int m = 16; m >= 1; m >>= 1) v += __shfl_xor(v, m, 32);
  return v;
}

// ---------------- weight prep: f32 [B,R,C] -> bf16 transposed [B,C,R] ----------------
__global__ void transpose_bf16(const float* __restrict__ src, __bf16* __restrict__ dst,
                               int R, int C, int B) {
  int total = B * R * C;
  for (int i = blockIdx.x * blockDim.x + threadIdx.x; i < total; i += gridDim.x * blockDim.x) {
    int b = i / (R * C); int rem = i - b * (R * C);
    int r = rem / C, c = rem - r * C;
    dst[b * R * C + c * R + r] = (__bf16)src[i];
  }
}

// ---------------- node incidence counts ----------------
__global__ void cnt_build(const int* __restrict__ hedges, float* __restrict__ cnt) {
  int i = blockIdx.x * 256 + threadIdx.x;
  if (i < EE * SSZ) atomicAdd(&cnt[hedges[i]], 1.0f);
}
__global__ void cnt_fin(float* cnt) {
  int i = threadIdx.x;
  if (i < NN) cnt[i] = fmaxf(cnt[i], 1.0f);
  else if (i < MP) cnt[i] = 1.0f;
}

// ---------------- input projection: h = gelu(LN(x@Wi+bi)) + PE ----------------
__global__ __launch_bounds__(256) void input_proj(
    const float* __restrict__ x, const float* __restrict__ Wi, const float* __restrict__ bi,
    const float* __restrict__ g, const float* __restrict__ b,
    float* __restrict__ h, __bf16* __restrict__ hbf) {
  int wave = threadIdx.x >> 5, lane = threadIdx.x & 31;
  int n = blockIdx.x * 8 + wave;
  if (n >= NN) return;
  float xr[DD];
  #pragma unroll
  for (int d = 0; d < DD; d++) xr[d] = x[n * DD + d];
  float y[4];
  #pragma unroll
  for (int cv = 0; cv < 4; cv++) {
    int ch = lane + cv * 32;
    float acc = bi[ch];
    #pragma unroll
    for (int d = 0; d < DD; d++) acc += xr[d] * Wi[d * HH + ch];
    y[cv] = acc;
  }
  float mean = wred_sum(y[0] + y[1] + y[2] + y[3]) * (1.0f / HH);
  float sq = 0.f;
  #pragma unroll
  for (int cv = 0; cv < 4; cv++) { float d = y[cv] - mean; sq += d * d; }
  float inv = rsqrtf(wred_sum(sq) * (1.0f / HH) + 1e-5f);
  const float pfac = 0.071955784175f;   // ln(10000)/128
  #pragma unroll
  for (int cv = 0; cv < 4; cv++) {
    int ch = lane + cv * 32;
    float v = geluf((y[cv] - mean) * inv * g[ch] + b[ch]);
    float dv = __expf(-(float)(ch & ~1) * pfac);
    float a = (float)n * dv;
    v += (ch & 1) ? cosf(a) : sinf(a);
    h[n * HH + ch] = v;
    hbf[n * HH + ch] = (__bf16)v;
  }
}

// ---------------- WMMA GEMM: C[M,N] = act(A_bf16[M,K] @ WT_bf16[N,K]^T + bias) ----------------
template <bool GELU, bool OUTBF>
__global__ __launch_bounds__(256) void gemm_bf16(
    const __bf16* __restrict__ A, const __bf16* __restrict__ WT,
    const float* __restrict__ bias, void* __restrict__ Cout, int Mt, int Ncols, int K) {
  int wave = blockIdx.x * 8 + (threadIdx.x >> 5);
  int lane = threadIdx.x & 31;
  int Nt = Ncols >> 4;
  if (wave >= Mt * Nt) return;
  int mt = wave / Nt, nt = wave - mt * Nt;
  int lm = lane & 15, hi = lane >> 4;
  int row = mt * 16 + lm;
  int col = nt * 16 + lm;
  int kb = hi * 8;
  v8f acc = {0.f, 0.f, 0.f, 0.f, 0.f, 0.f, 0.f, 0.f};
  #pragma unroll 4
  for (int k0 = 0; k0 < K; k0 += 32) {
    const __bf16* ap = A + row * K + k0 + kb;       // A layout: K kb..kb+7, 16+kb..+7
    const __bf16* bp = WT + col * K + k0 + kb;      // B layout: K kb..kb+7, 16+kb..+7
    v16bf au, bu;
    #pragma unroll
    for (int j = 0; j < 8; j++) { au[j] = ap[j]; au[8 + j] = ap[16 + j]; }
    #pragma unroll
    for (int j = 0; j < 8; j++) { bu[j] = bp[j]; bu[8 + j] = bp[16 + j]; }
    acc = __builtin_amdgcn_wmma_f32_16x16x32_bf16(false, au, false, bu,
                                                  (short)0, acc, false, false);
  }
  float bv = bias[col];
  int rbase = mt * 16 + hi * 8;
  #pragma unroll
  for (int r = 0; r < 8; r++) {
    float v = acc[r] + bv;
    if (GELU) v = geluf(v);
    if (OUTBF) ((__bf16*)Cout)[(rbase + r) * Ncols + col] = (__bf16)v;
    else       ((float*)Cout)[(rbase + r) * Ncols + col] = v;
  }
}

// ---------------- hypergraph conv: ef=mean_s c[hedges], scatter-add (LDS-accumulated) ----------------
__global__ __launch_bounds__(256) void conv_scatter(
    const float* __restrict__ c, const int* __restrict__ hedges,
    float* __restrict__ accg, int epg) {
  __shared__ float lacc[NN * HH];   // 100 KB of the 320 KB/WGP LDS
  for (int i = threadIdx.x; i < NN * HH; i += 256) lacc[i] = 0.f;
  __syncthreads();
  int wave = threadIdx.x >> 5, lane = threadIdx.x & 31;
  int e0 = blockIdx.x * epg;
  for (int e = e0 + wave; e < e0 + epg; e += 8) {
    int nodes[SSZ];
    #pragma unroll
    for (int s = 0; s < SSZ; s++) nodes[s] = hedges[e * SSZ + s];
    #pragma unroll
    for (int cv = 0; cv < 4; cv++) {
      int ch = lane + cv * 32;
      float sum = 0.f;
      #pragma unroll
      for (int s = 0; s < SSZ; s++) sum += c[nodes[s] * HH + ch];
      sum *= 0.125f;
      #pragma unroll
      for (int s = 0; s < SSZ; s++) atomicAdd(&lacc[nodes[s] * HH + ch], sum);
    }
  }
  __syncthreads();
  for (int i = threadIdx.x; i < NN * HH; i += 256) {
    float v = lacc[i];
    if (v != 0.f) atomicAdd(&accg[i], v);
  }
}

// ---------------- per-edge MHA, 2 edges packed per 16x16 WMMA tile ----------------
// Softmax done in A-layout: each lane owns 8 consecutive key-columns of one row,
// so max/exp/sum are purely in-lane (no cross-lane shuffles, no ds_bpermute).
__global__ __launch_bounds__(256) void attn_edges(
    const float* __restrict__ q, const float* __restrict__ k, const float* __restrict__ v,
    const int* __restrict__ hedges, float* __restrict__ aggg, int ppg) {
  __shared__ float lagg[NN * HH];
  __shared__ float stg[8][16][17];
  __shared__ int   nstg[8][16];
  for (int i = threadIdx.x; i < NN * HH; i += 256) lagg[i] = 0.f;
  __syncthreads();
  int wave = threadIdx.x >> 5, lane = threadIdx.x & 31;
  int lm = lane & 15, hi = lane >> 4;
  int kb = hi * 8;
  const float scale = 0.17677669529663689f;  // 1/sqrt(32)
  // A-fragment lane validity for the block-diagonal (2 edges per tile) mask:
  bool lv = ((lm >> 3) == hi);
  int p0 = blockIdx.x * ppg;
  for (int p = p0 + wave; p < p0 + ppg; p += 8) {
    int fbase = p * 16;                       // rows 0..15 = edges (2p, 2p+1), s 0..7
    int nodeL = hedges[fbase + lm];           // node for my row/col index lm
    __builtin_amdgcn_wave_barrier();          // prev iter's nstg reads complete (in-order DS)
    nstg[wave][lm] = nodeL;
    __builtin_amdgcn_wave_barrier();
    const float* qrow = q + nodeL * HH;
    const float* krow = k + nodeL * HH;
    #pragma unroll 1
    for (int hd = 0; hd < 4; hd++) {
      int base = hd * 32;
      // A = Q fragment (16x32 bf16), B = K^T fragment (32x16 bf16)
      v16bf au, bu;
      #pragma unroll
      for (int j = 0; j < 8; j++) {
        au[j]     = (__bf16)qrow[base + kb + j];
        au[8 + j] = (__bf16)qrow[base + 16 + kb + j];
      }
      #pragma unroll
      for (int j = 0; j < 8; j++) { bu[j] = (__bf16)krow[base + kb + j]; bu[8 + j] = (__bf16)krow[base + 16 + kb + j]; }
      v8f sc = {0.f, 0.f, 0.f, 0.f, 0.f, 0.f, 0.f, 0.f};
      sc = __builtin_amdgcn_wmma_f32_16x16x32_bf16(false, au, false, bu,
                                                   (short)0, sc, false, false);
      // stage raw scores (C layout) -> LDS
      #pragma unroll
      for (int r = 0; r < 8; r++) stg[wave][r + 8 * hi][lm] = sc[r];
      __builtin_amdgcn_wave_barrier();
      // reload in A layout: lane owns row lm, key-cols kb..kb+7 -> in-lane softmax
      float sv[8];
      #pragma unroll
      for (int j = 0; j < 8; j++) sv[j] = stg[wave][lm][kb + j] * scale;
      __builtin_amdgcn_wave_barrier();
      float mx = sv[0];
      #pragma unroll
      for (int j = 1; j < 8; j++) mx = fmaxf(mx, sv[j]);
      float sum = 0.f;
      #pragma unroll
      for (int j = 0; j < 8; j++) { sv[j] = __expf(sv[j] - mx); sum += sv[j]; }
      float rs = 1.0f / sum;
      v16bf pf;
      #pragma unroll
      for (int j = 0; j < 8; j++) {
        pf[j]     = lv ? (__bf16)(sv[j] * rs) : (__bf16)0.f;  // zero cross-edge block
        pf[8 + j] = (__bf16)0.f;                              // K 16..31 zero pad
      }
      // out[16x32] = attn[16x16(pad32)] @ V[16x32], two 16-col halves
      #pragma unroll
      for (int half = 0; half < 2; half++) {
        v16bf vu;
        // K rows kb..kb+7 = v rows; K 16..31 zero pad
        #pragma unroll
        for (int j = 0; j < 8; j++) {
          vu[j]     = (__bf16)v[nstg[wave][kb + j] * HH + base + half * 16 + lm];
          vu[8 + j] = (__bf16)0.f;
        }
        v8f oc = {0.f, 0.f, 0.f, 0.f, 0.f, 0.f, 0.f, 0.f};
        oc = __builtin_amdgcn_wmma_f32_16x16x32_bf16(false, pf, false, vu,
                                                     (short)0, oc, false, false);
        #pragma unroll
        for (int r = 0; r < 8; r++)
          atomicAdd(&lagg[nstg[wave][r + 8 * hi] * HH + base + half * 16 + lm], oc[r]);
      }
    }
  }
  __syncthreads();
  for (int i = threadIdx.x; i < NN * HH; i += 256) {
    float vv = lagg[i];
    if (vv != 0.f) atomicAdd(&aggg[i], vv);
  }
}

// ---------------- h = LN(h + add[*1/cnt]); writes f32 + bf16 ----------------
__global__ __launch_bounds__(256) void residual_ln(
    float* __restrict__ h, __bf16* __restrict__ hbf,
    const float* __restrict__ add, const float* __restrict__ cnt,
    const float* __restrict__ g, const float* __restrict__ b) {
  int wave = threadIdx.x >> 5, lane = threadIdx.x & 31;
  int n = blockIdx.x * 8 + wave;
  if (n >= NN) return;
  float ic = cnt ? (1.0f / cnt[n]) : 1.0f;
  float y[4];
  #pragma unroll
  for (int cv = 0; cv < 4; cv++) {
    int ch = lane + cv * 32;
    y[cv] = h[n * HH + ch] + add[n * HH + ch] * ic;
  }
  float mean = wred_sum(y[0] + y[1] + y[2] + y[3]) * (1.0f / HH);
  float sq = 0.f;
  #pragma unroll
  for (int cv = 0; cv < 4; cv++) { float d = y[cv] - mean; sq += d * d; }
  float inv = rsqrtf(wred_sum(sq) * (1.0f / HH) + 1e-5f);
  #pragma unroll
  for (int cv = 0; cv < 4; cv++) {
    int ch = lane + cv * 32;
    float v = (y[cv] - mean) * inv * g[ch] + b[ch];
    h[n * HH + ch] = v;
    hbf[n * HH + ch] = (__bf16)v;
  }
}

__global__ __launch_bounds__(256) void add_writebf(
    float* __restrict__ h, __bf16* __restrict__ hbf, const float* __restrict__ ho) {
  int wave = threadIdx.x >> 5, lane = threadIdx.x & 31;
  int n = blockIdx.x * 8 + wave;
  if (n >= NN) return;
  #pragma unroll
  for (int cv = 0; cv < 4; cv++) {
    int ch = lane + cv * 32;
    float v = h[n * HH + ch] + ho[n * HH + ch];
    h[n * HH + ch] = v;
    hbf[n * HH + ch] = (__bf16)v;
  }
}

__global__ __launch_bounds__(256) void scale_bf(
    const float* __restrict__ agg, const float* __restrict__ cnt, __bf16* __restrict__ ab) {
  int wave = threadIdx.x >> 5, lane = threadIdx.x & 31;
  int n = blockIdx.x * 8 + wave;
  if (n >= NN) return;
  float ic = 1.0f / cnt[n];
  #pragma unroll
  for (int cv = 0; cv < 4; cv++) {
    int ch = lane + cv * 32;
    ab[n * HH + ch] = (__bf16)(agg[n * HH + ch] * ic);
  }
}

// ---------------- pooling + MLP heads + outputs ----------------
__global__ __launch_bounds__(384) void pool_head(
    const float* __restrict__ h,
    const float* __restrict__ Wh1, const float* __restrict__ bh1,
    const float* __restrict__ Wh2, const float* __restrict__ bh2,
    const float* __restrict__ Wh3, const float* __restrict__ bh3,
    const float* __restrict__ We,  const float* __restrict__ be,
    float* __restrict__ out) {
  __shared__ float g[384];
  __shared__ float t1[3][128];
  __shared__ float t2[3][64];
  __shared__ float t3[3];
  int t = threadIdx.x;
  if (t < 128) {
    float sum = 0.f, mx = -3.0e38f;
    for (int r = 0; r < NN; r++) { float v = h[r * HH + t]; sum += v; mx = fmaxf(mx, v); }
    g[t] = sum / (float)NN; g[128 + t] = mx; g[256 + t] = sum;
  }
  __syncthreads();
  {
    int kk = t >> 7, o = t & 127;
    float acc = bh1[kk * 128 + o];
    for (int i = 0; i < 384; i++) acc += g[i] * Wh1[kk * 384 * 128 + i * 128 + o];
    t1[kk][o] = geluf(acc);
  }
  __syncthreads();
  if (t < 192) {
    int kk = t >> 6, o = t & 63;
    float acc = bh2[kk * 64 + o];
    for (int i = 0; i < 128; i++) acc += t1[kk][i] * Wh2[kk * 128 * 64 + i * 64 + o];
    t2[kk][o] = geluf(acc);
  }
  __syncthreads();
  if (t < 3) {
    float acc = bh3[t];
    for (int i = 0; i < 64; i++) acc += t2[t][i] * Wh3[t * 64 + i];
    t3[t] = acc;
  }
  __syncthreads();
  if (t == 0) {
    float d0 = t3[0]; out[0] = (d0 > 20.f ? d0 : log1pf(__expf(d0))) + 1.0f;
    out[1] = 1.0f / (1.0f + __expf(-t3[1]));
    float d2 = t3[2]; out[2] = (d2 > 20.f ? d2 : log1pf(__expf(d2)));
  }
  if (t < 128) {
    float acc = be[t];
    for (int i = 0; i < 384; i++) acc += g[i] * We[i * 128 + t];
    out[3 + t] = acc;
  }
}

// ==================================================================================
extern "C" void kernel_launch(void* const* d_in, const int* in_sizes, int n_in,
                              void* d_out, int out_size, void* d_ws, size_t ws_size,
                              hipStream_t stream) {
  (void)in_sizes; (void)n_in; (void)out_size; (void)ws_size;
  const float* x      = (const float*)d_in[0];
  const int*   hedges = (const int*)  d_in[1];
  const float* Wi  = (const float*)d_in[2];  const float* bi  = (const float*)d_in[3];
  const float* lnig= (const float*)d_in[4];  const float* lnib= (const float*)d_in[5];
  const float* Wc  = (const float*)d_in[6];  const float* bc  = (const float*)d_in[7];
  const float* Wq  = (const float*)d_in[8];  const float* bq  = (const float*)d_in[9];
  const float* Wk  = (const float*)d_in[10]; const float* bk  = (const float*)d_in[11];
  const float* Wv  = (const float*)d_in[12]; const float* bv  = (const float*)d_in[13];
  const float* Wo  = (const float*)d_in[14]; const float* bo  = (const float*)d_in[15];
  const float* n1g = (const float*)d_in[16]; const float* n1b = (const float*)d_in[17];
  const float* n2g = (const float*)d_in[18]; const float* n2b = (const float*)d_in[19];
  const float* Wf1 = (const float*)d_in[20]; const float* bf1 = (const float*)d_in[21];
  const float* Wf2 = (const float*)d_in[22]; const float* bf2 = (const float*)d_in[23];
  const float* Wh1 = (const float*)d_in[24]; const float* bh1 = (const float*)d_in[25];
  const float* Wh2 = (const float*)d_in[26]; const float* bh2 = (const float*)d_in[27];
  const float* Wh3 = (const float*)d_in[28]; const float* bh3 = (const float*)d_in[29];
  const float* We  = (const float*)d_in[30]; const float* be  = (const float*)d_in[31];

  char* base = (char*)d_ws;
  size_t off = 0;
  auto carve = [&](size_t bytes) -> void* {
    void* p = base + off; off += (bytes + 255) & ~(size_t)255; return p;
  };
  __bf16* WcT  = (__bf16*)carve(LL * HH * HH * 2);
  __bf16* WqT  = (__bf16*)carve(LL * HH * HH * 2);
  __bf16* WkT  = (__bf16*)carve(LL * HH * HH * 2);
  __bf16* WvT  = (__bf16*)carve(LL * HH * HH * 2);
  __bf16* WoT  = (__bf16*)carve(LL * HH * HH * 2);
  __bf16* Wf1T = (__bf16*)carve(LL * FFD * HH * 2);
  __bf16* Wf2T = (__bf16*)carve(LL * HH * FFD * 2);
  char* actBase = base + off;
  float*  hbuf  = (float*)carve(MP * HH * 4);
  __bf16* hbf   = (__bf16*)carve(MP * HH * 2);
  float*  cbuf  = (float*)carve(MP * HH * 4);
  float*  qbuf  = (float*)carve(MP * HH * 4);
  float*  kbuf  = (float*)carve(MP * HH * 4);
  float*  vbuf  = (float*)carve(MP * HH * 4);
  float*  accb  = (float*)carve(MP * HH * 4);
  float*  aggf  = (float*)carve(MP * HH * 4);
  __bf16* aggbf = (__bf16*)carve(MP * HH * 2);
  float*  hob   = (float*)carve(MP * HH * 4);
  __bf16* midb  = (__bf16*)carve(MP * FFD * 2);
  float*  f2b   = (float*)carve(MP * HH * 4);
  float*  cntb  = (float*)carve(MP * 4);
  size_t actBytes = (size_t)((base + off) - actBase);

  // weight prep (bf16, transposed to [N,K])
  transpose_bf16<<<256, 256, 0, stream>>>(Wc,  WcT,  HH,  HH,  LL);
  transpose_bf16<<<256, 256, 0, stream>>>(Wq,  WqT,  HH,  HH,  LL);
  transpose_bf16<<<256, 256, 0, stream>>>(Wk,  WkT,  HH,  HH,  LL);
  transpose_bf16<<<256, 256, 0, stream>>>(Wv,  WvT,  HH,  HH,  LL);
  transpose_bf16<<<256, 256, 0, stream>>>(Wo,  WoT,  HH,  HH,  LL);
  transpose_bf16<<<256, 256, 0, stream>>>(Wf1, Wf1T, HH,  FFD, LL);
  transpose_bf16<<<256, 256, 0, stream>>>(Wf2, Wf2T, FFD, HH,  LL);

  hipMemsetAsync(actBase, 0, actBytes, stream);
  cnt_build<<<(EE * SSZ) / 256, 256, 0, stream>>>(hedges, cntb);
  cnt_fin<<<1, 256, 0, stream>>>(cntb);
  input_proj<<<25, 256, 0, stream>>>(x, Wi, bi, lnig, lnib, hbuf, hbf);

  for (int l = 0; l < LL; l++) {
    hipMemsetAsync(accb, 0, MP * HH * 4, stream);
    hipMemsetAsync(aggf, 0, MP * HH * 4, stream);
    // hypergraph conv
    gemm_bf16<false, false><<<13, 256, 0, stream>>>(hbf, WcT + l * HH * HH, bc + l * HH, cbuf, MTILES, HH, HH);
    conv_scatter<<<64, 256, 0, stream>>>(cbuf, hedges, accb, EE / 64);
    residual_ln<<<25, 256, 0, stream>>>(hbuf, hbf, accb, cntb, n1g + l * HH, n1b + l * HH);
    // attention
    gemm_bf16<false, false><<<13, 256, 0, stream>>>(hbf, WqT + l * HH * HH, bq + l * HH, qbuf, MTILES, HH, HH);
    gemm_bf16<false, false><<<13, 256, 0, stream>>>(hbf, WkT + l * HH * HH, bk + l * HH, kbuf, MTILES, HH, HH);
    gemm_bf16<false, false><<<13, 256, 0, stream>>>(hbf, WvT + l * HH * HH, bv + l * HH, vbuf, MTILES, HH, HH);
    attn_edges<<<64, 256, 0, stream>>>(qbuf, kbuf, vbuf, hedges, aggf, (EE / 2) / 64);
    scale_bf<<<25, 256, 0, stream>>>(aggf, cntb, aggbf);
    gemm_bf16<false, false><<<13, 256, 0, stream>>>(aggbf, WoT + l * HH * HH, bo + l * HH, hob, MTILES, HH, HH);
    add_writebf<<<25, 256, 0, stream>>>(hbuf, hbf, hob);
    // FFN
    gemm_bf16<true, true><<<52, 256, 0, stream>>>(hbf, Wf1T + l * FFD * HH, bf1 + l * FFD, midb, MTILES, FFD, HH);
    gemm_bf16<false, false><<<13, 256, 0, stream>>>(midb, Wf2T + l * HH * FFD, bf2 + l * HH, f2b, MTILES, HH, FFD);
    residual_ln<<<25, 256, 0, stream>>>(hbuf, hbf, f2b, nullptr, n2g + l * HH, n2b + l * HH);
  }
  pool_head<<<1, 384, 0, stream>>>(hbuf, Wh1, bh1, Wh2, bh2, Wh3, bh3, We, be, (float*)d_out);
}
